// LstmAggregator_48077863911549
// MI455X (gfx1250) — compile-verified, matrix operand order checked by reading, not run on hardware
//
#include <hip/hip_runtime.h>
#include <hip/hip_bf16.h>

typedef _Float16 v8h  __attribute__((ext_vector_type(8)));
typedef _Float16 v16h __attribute__((ext_vector_type(16)));
typedef float    v8f  __attribute__((ext_vector_type(8)));
typedef float    v4f  __attribute__((ext_vector_type(4)));

#define FDIM 128
#define HDIM 256
#define KTOT 384          // F + H
#define GATES 1024        // 4*H
#define OUTD 256
#define ASTR 392          // padded LDS row stride in halves (392*2=784 B, bank-spread)
#define CS   2048         // edge chunk size for stable-rank build
#define LTHREADS 512      // 16 wave32s per 32-node tile

union V16U { v16h v; v8h h[2]; };

__device__ inline float sig_f(float x)  { return 1.f / (1.f + __expf(-x)); }
__device__ inline float tanh_f(float x) { float e = __expf(-2.f * x); return (1.f - e) / (1.f + e); }

__device__ inline v16h load_frag16(const _Float16* p) {
    V16U u;
    u.h[0] = *(const v8h*)p;
    u.h[1] = *(const v8h*)(p + 16);
    return u.v;
}

// async 16B global->LDS copy (gfx1250 LDS-DMA, ASYNCcnt-tracked)
__device__ inline void async_copy_b128(const _Float16* gsrc, _Float16* ldst) {
    unsigned int loff = (unsigned int)(unsigned long long)ldst;   // truncated generic ptr == LDS byte offset
    unsigned long long gp = (unsigned long long)gsrc;
    asm volatile("global_load_async_to_lds_b128 %0, %1, off"
                 :: "v"(loff), "v"(gp) : "memory");
}
__device__ inline void wait_async0() {
    asm volatile("s_wait_asynccnt 0x0" ::: "memory");
}

// ---------------- CSR construction (deterministic, matches stable argsort) ----------------

__global__ void zero_i32_kernel(int* p, size_t n) {
    size_t i = (size_t)blockIdx.x * blockDim.x + threadIdx.x;
    for (; i < n; i += (size_t)gridDim.x * blockDim.x) p[i] = 0;
}

// per-chunk in-order rank + chunk histogram
__global__ void chunk_rank_kernel(const int* __restrict__ src, int* __restrict__ r_in,
                                  int* __restrict__ Hh, int E_, int N_) {
    __shared__ int s[CS];
    int c = blockIdx.x;
    int base = c * CS;
    for (int i = threadIdx.x; i < CS; i += blockDim.x) {
        int e = base + i;
        s[i] = (e < E_) ? src[e] : -1;
    }
    __syncthreads();
    for (int i = threadIdx.x; i < CS; i += blockDim.x) {
        int e = base + i;
        if (e >= E_) continue;
        int my = s[i];
        int r = 0;
        for (int j = 0; j < i; ++j) r += (s[j] == my) ? 1 : 0;
        r_in[e] = r;
        atomicAdd(&Hh[(size_t)c * N_ + my], 1);
    }
}

// turn per-chunk histograms into per-node chunk prefixes; emit total counts
__global__ void chunk_prefix_kernel(int* __restrict__ Hh, int* __restrict__ counts,
                                    int N_, int nchunks) {
    int n = blockIdx.x * blockDim.x + threadIdx.x;
    if (n >= N_) return;
    int run = 0;
    for (int c = 0; c < nchunks; ++c) {
        size_t idx = (size_t)c * N_ + n;
        int t = Hh[idx];
        Hh[idx] = run;
        run += t;
    }
    counts[n] = run;
}

// single-block exclusive scan of counts -> offsets
__global__ void scan_kernel(const int* __restrict__ counts, int* __restrict__ offsets, int n) {
    __shared__ int buf[1024];
    __shared__ int base_s;
    if (threadIdx.x == 0) base_s = 0;
    __syncthreads();
    for (int start = 0; start < n; start += 1024) {
        int i = start + threadIdx.x;
        int v = (i < n) ? counts[i] : 0;
        buf[threadIdx.x] = v;
        __syncthreads();
        for (int off = 1; off < 1024; off <<= 1) {
            int t = (threadIdx.x >= off) ? buf[threadIdx.x - off] : 0;
            __syncthreads();
            buf[threadIdx.x] += t;
            __syncthreads();
        }
        int incl = buf[threadIdx.x];
        if (i < n) offsets[i] = base_s + (incl - v);
        __syncthreads();
        if (threadIdx.x == 1023) base_s += incl;
        __syncthreads();
    }
}

__global__ void fill_kernel(const int* __restrict__ src, const int* __restrict__ trg,
                            const int* __restrict__ r_in, const int* __restrict__ Hh,
                            const int* __restrict__ offsets, int* __restrict__ nbr,
                            int E_, int N_) {
    int e = blockIdx.x * blockDim.x + threadIdx.x;
    if (e >= E_) return;
    int c = e / CS;
    int n = src[e];
    int slot = offsets[n] + Hh[(size_t)c * N_ + n] + r_in[e];
    nbr[slot] = trg[e];
}

// ---------------- weight + feature conversion to f16 ----------------

__global__ void convert_kernel(const float* __restrict__ Wih, const float* __restrict__ Whh,
                               const float* __restrict__ bih, const float* __restrict__ bhh,
                               const float* __restrict__ Wout, const float* __restrict__ X,
                               _Float16* __restrict__ Wcat, _Float16* __restrict__ WoutT,
                               float* __restrict__ bias, _Float16* __restrict__ Xh, int N_) {
    const int tot1 = GATES * KTOT;     // Wcat
    const int tot2 = OUTD * KTOT;      // WoutT
    const int tot3 = GATES;            // bias
    const int tot4 = N_ * FDIM;        // Xh
    int stride = gridDim.x * blockDim.x;
    for (int idx = blockIdx.x * blockDim.x + threadIdx.x;
         idx < tot1 + tot2 + tot3 + tot4; idx += stride) {
        if (idx < tot1) {
            int r = idx / KTOT, k = idx - r * KTOT;
            float v = (k < FDIM) ? Wih[r * FDIM + k] : Whh[r * HDIM + (k - FDIM)];
            Wcat[idx] = (_Float16)v;
        } else if (idx < tot1 + tot2) {
            int j = idx - tot1;
            int n = j / KTOT, k = j - n * KTOT;
            WoutT[j] = (_Float16)Wout[k * OUTD + n];
        } else if (idx < tot1 + tot2 + tot3) {
            int i = idx - tot1 - tot2;
            bias[i] = bih[i] + bhh[i];
        } else {
            int i = idx - tot1 - tot2 - tot3;
            Xh[i] = (_Float16)X[i];
        }
    }
}

// ---------------- main LSTM + fused output GEMM ----------------
// 1 workgroup (16 wave32s) = 32 nodes.
// Wave w: row-tile rt = w&1 (16 nodes), hidden-column group wg = w>>1 (cols wg*32..wg*32+31).

__global__ void __launch_bounds__(LTHREADS, 1)
lstm_wmma_kernel(const _Float16* __restrict__ Xh, const int* __restrict__ counts,
                 const int* __restrict__ offsets, const int* __restrict__ nbr,
                 const _Float16* __restrict__ Wcat, const float* __restrict__ bias,
                 const _Float16* __restrict__ WoutT, float* __restrict__ out, int N_) {
    __shared__ _Float16 A[32 * ASTR];   // [row][k]: k<128 = x_t, k in [128,384) = h
    __shared__ int cnt_s[32];
    __shared__ int off_s[32];
    __shared__ int tmax_s;

    const int tid = threadIdx.x;
    const int lane = tid & 31;
    const int w = tid >> 5;             // wave 0..15
    const int rt = w & 1;               // row tile (nodes rt*16 .. rt*16+15)
    const int wg = w >> 1;              // hidden column group 0..7
    const int half = lane >> 4;         // lane 16..31 -> second half
    const int nl = lane & 15;
    const int node0 = blockIdx.x * 32;

    if (tid < 32) {
        int nd = node0 + tid;
        cnt_s[tid] = (nd < N_) ? counts[nd] : 0;
        off_s[tid] = (nd < N_) ? offsets[nd] : 0;
    }
    for (int i = tid; i < 32 * ASTR; i += LTHREADS) A[i] = (_Float16)0.f;
    __syncthreads();
    if (tid == 0) {
        int m = 0;
        for (int i = 0; i < 32; ++i) m = (cnt_s[i] > m) ? cnt_s[i] : m;
        tmax_s = m;
    }
    __syncthreads();
    const int tmax = tmax_s;

    // per-lane constants
    float breg[4][2];
    #pragma unroll
    for (int g = 0; g < 4; ++g)
        #pragma unroll
        for (int ct = 0; ct < 2; ++ct)
            breg[g][ct] = bias[g * HDIM + wg * 32 + ct * 16 + nl];

    int creg[8];
    #pragma unroll
    for (int k = 0; k < 8; ++k)
        creg[k] = cnt_s[rt * 16 + half * 8 + k];

    v8f cst[2];                          // cell state per ct
    #pragma unroll
    for (int ct = 0; ct < 2; ++ct)
        #pragma unroll
        for (int k = 0; k < 8; ++k) cst[ct][k] = 0.f;

    const int gr = tid >> 4;            // gather row 0..31
    const int gseg = tid & 15;          // 16-byte (8-half) segment within the row

    for (int t = 0; t < tmax; ++t) {
        // phase A: gather neighbor features (f16) into A[:,0:128] via async LDS-DMA
        {
            _Float16* dst = &A[gr * ASTR + gseg * 8];
            if (t < cnt_s[gr]) {
                int nb = nbr[off_s[gr] + t];
                async_copy_b128(Xh + (size_t)nb * FDIM + gseg * 8, dst);
            } else {
                *(v8h*)dst = (v8h)((_Float16)0.f);
            }
        }
        wait_async0();
        __syncthreads();  // prev-step h stores + x copies visible

        // phase B: gates = [x_t | h] @ Wcat^T + bias, 96 WMMAs/wave
        v8f acc[4][2]; // [gate][ct]
        #pragma unroll
        for (int g = 0; g < 4; ++g)
            #pragma unroll
            for (int ct = 0; ct < 2; ++ct)
                #pragma unroll
                for (int k = 0; k < 8; ++k) acc[g][ct][k] = breg[g][ct];

        #pragma unroll 2
        for (int kt = 0; kt < 12; ++kt) {
            int kb = kt * 32;
            v16h a = load_frag16(&A[(rt * 16 + nl) * ASTR + kb + half * 8]);
            #pragma unroll
            for (int g = 0; g < 4; ++g) {
                #pragma unroll
                for (int ct = 0; ct < 2; ++ct) {
                    const _Float16* bp =
                        Wcat + (size_t)(g * HDIM + wg * 32 + ct * 16 + nl) * KTOT + kb + half * 8;
                    v16h b = load_frag16(bp);
                    acc[g][ct] = __builtin_amdgcn_wmma_f32_16x16x32_f16(
                        false, a, false, b, (short)0, acc[g][ct], false, false);
                }
            }
        }
        __syncthreads();  // all LDS A reads done before rewrites

        // phase C: elementwise LSTM update; masked write-back of h (f16) into A[:,128:384]
        #pragma unroll
        for (int ct = 0; ct < 2; ++ct) {
            #pragma unroll
            for (int k = 0; k < 8; ++k) {
                float i_ = sig_f(acc[0][ct][k]);
                float f_ = sig_f(acc[1][ct][k]);
                float g_ = tanh_f(acc[2][ct][k]);
                float o_ = sig_f(acc[3][ct][k]);
                float cn = f_ * cst[ct][k] + i_ * g_;
                float hn = o_ * tanh_f(cn);
                if (t < creg[k]) {
                    cst[ct][k] = cn;
                    int row = rt * 16 + half * 8 + k;
                    A[row * ASTR + FDIM + wg * 32 + ct * 16 + nl] = (_Float16)hn;
                }
            }
        }
    }

    // ---- fused output GEMM: out = [x_own | h] @ W_out ----
    __syncthreads();
    {
        int nd = node0 + gr;
        _Float16* dst = &A[gr * ASTR + gseg * 8];
        if (nd < N_) {
            async_copy_b128(Xh + (size_t)nd * FDIM + gseg * 8, dst);
        } else {
            *(v8h*)dst = (v8h)((_Float16)0.f);
        }
    }
    wait_async0();
    __syncthreads();

    v8f oacc[2]; // [ct]
    #pragma unroll
    for (int ct = 0; ct < 2; ++ct)
        #pragma unroll
        for (int k = 0; k < 8; ++k) oacc[ct][k] = 0.f;

    #pragma unroll 2
    for (int kt = 0; kt < 12; ++kt) {
        int kb = kt * 32;
        v16h a = load_frag16(&A[(rt * 16 + nl) * ASTR + kb + half * 8]);
        #pragma unroll
        for (int ct = 0; ct < 2; ++ct) {
            const _Float16* bp =
                WoutT + (size_t)(wg * 32 + ct * 16 + nl) * KTOT + kb + half * 8;
            v16h b = load_frag16(bp);
            oacc[ct] = __builtin_amdgcn_wmma_f32_16x16x32_f16(
                false, a, false, b, (short)0, oacc[ct], false, false);
        }
    }

    #pragma unroll
    for (int ct = 0; ct < 2; ++ct)
        #pragma unroll
        for (int k = 0; k < 8; ++k) {
            int row = rt * 16 + half * 8 + k;
            int nd = node0 + row;
            if (nd < N_) out[(size_t)nd * OUTD + wg * 32 + ct * 16 + nl] = oacc[ct][k];
        }
}

// ---------------- host launch ----------------

extern "C" void kernel_launch(void* const* d_in, const int* in_sizes, int n_in,
                              void* d_out, int out_size, void* d_ws, size_t ws_size,
                              hipStream_t stream) {
    const float* X    = (const float*)d_in[0];
    const int*   adj  = (const int*)d_in[1];
    const float* Wih  = (const float*)d_in[2];
    const float* Whh  = (const float*)d_in[3];
    const float* bih  = (const float*)d_in[4];
    const float* bhh  = (const float*)d_in[5];
    const float* Wout = (const float*)d_in[6];
    float* out = (float*)d_out;

    const int N_ = in_sizes[0] / FDIM;
    const int E_ = in_sizes[1] / 2;
    const int nchunks = (E_ + CS - 1) / CS;
    const int* src = adj;
    const int* trg = adj + E_;

    // carve workspace (256B aligned)
    char* p = (char*)d_ws;
    auto carve = [&](size_t bytes) -> void* {
        void* r = (void*)p;
        p += (bytes + 255) & ~(size_t)255;
        return r;
    };
    _Float16* Wcat  = (_Float16*)carve((size_t)GATES * KTOT * sizeof(_Float16));
    _Float16* WoutT = (_Float16*)carve((size_t)OUTD * KTOT * sizeof(_Float16));
    float*    bias  = (float*)carve((size_t)GATES * sizeof(float));
    _Float16* Xh    = (_Float16*)carve((size_t)N_ * FDIM * sizeof(_Float16));
    int*      cnts  = (int*)carve((size_t)N_ * sizeof(int));
    int*      offs  = (int*)carve((size_t)N_ * sizeof(int));
    int*      nbr   = (int*)carve((size_t)E_ * sizeof(int));
    int*      r_in  = (int*)carve((size_t)E_ * sizeof(int));
    int*      Hh    = (int*)carve((size_t)nchunks * N_ * sizeof(int));

    size_t histN = (size_t)nchunks * N_;
    int zgrid = (int)((histN + 255) / 256);
    if (zgrid > 8192) zgrid = 8192;
    zero_i32_kernel<<<zgrid, 256, 0, stream>>>(Hh, histN);
    chunk_rank_kernel<<<nchunks, 1024, 0, stream>>>(src, r_in, Hh, E_, N_);
    chunk_prefix_kernel<<<(N_ + 255) / 256, 256, 0, stream>>>(Hh, cnts, N_, nchunks);
    scan_kernel<<<1, 1024, 0, stream>>>(cnts, offs, N_);
    fill_kernel<<<(E_ + 255) / 256, 256, 0, stream>>>(src, trg, r_in, Hh, offs, nbr, E_, N_);
    convert_kernel<<<2048, 256, 0, stream>>>(Wih, Whh, bih, bhh, Wout, X, Wcat, WoutT, bias, Xh, N_);
    lstm_wmma_kernel<<<(N_ + 31) / 32, LTHREADS, 0, stream>>>(Xh, cnts, offs, nbr, Wcat, bias, WoutT, out, N_);
}
